// LlamaAttentionExperimental_41747082117643
// MI455X (gfx1250) — matrix operand order repaired and checked by reading, not compiled, not run on hardware
//
#include <hip/hip_runtime.h>
#include <cstddef>
#include <cstdint>

// ---------------------------------------------------------------------------
// Sparse GQA attention for MI455X (gfx1250), bf16 WMMA pipeline.
//   B=1, S=2048, HID=2048, H=16, KVH=8, D=128
// Roofline: dominated by the 256MB sorted_indices scan (~11us @ 23.3TB/s).
// GEMMs: v_wmma_f32_16x16x32_bf16 with async global->LDS double buffering.
// ---------------------------------------------------------------------------

#define S_LEN   2048
#define HID     2048
#define NH      16
#define KVH     8
#define HD      128           // head dim
#define HHD     (NH * HD)     // 2048
#define KVHD    (KVH * HD)    // 1024
#define WIN     128
#define MINIDX  4

typedef __attribute__((ext_vector_type(16))) __bf16       bfrag;
typedef __attribute__((ext_vector_type(8)))  float        v8f;
typedef __attribute__((ext_vector_type(4)))  unsigned int u32x4;

#define INFF __builtin_inff()

union FragU { bfrag v; u32x4 q[2]; };

// Build a 16x(bf16) fragment from two 16-byte chunks (works for LDS or global).
__device__ __forceinline__ bfrag load_frag2(const __bf16* p0, const __bf16* p1) {
    FragU f;
    f.q[0] = *(const u32x4*)p0;
    f.q[1] = *(const u32x4*)p1;
    return f.v;
}

// CDNA5 async copy: 16B per lane, global -> LDS, tracked by ASYNCcnt.
// (inline asm: portable across ROCm7.2 / amdgpu-toolchain builtin arity split)
__device__ __forceinline__ void async_ld_b128(unsigned lds_addr, const void* gaddr) {
    asm volatile("global_load_async_to_lds_b128 %0, %1, off"
                 :: "v"(lds_addr), "v"(gaddr) : "memory");
}
// generic LDS pointer -> 32-bit LDS address (flat aperture: addr[31:0] = LDS offset)
__device__ __forceinline__ unsigned lds_addr_of(const void* p) {
    return (unsigned)(uintptr_t)p;
}

// ---------------------------------------------------------------------------
// 1) elementwise f32 -> bf16
// ---------------------------------------------------------------------------
__global__ void cvt_bf16_kernel(const float* __restrict__ src,
                                __bf16* __restrict__ dst, int n) {
    int t = blockIdx.x * blockDim.x + threadIdx.x;
    if (t < n) dst[t] = (__bf16)src[t];
}

// src [R][C] f32 (row-major)  ->  dst [C][R] bf16 (row-major)
__global__ void transpose_cvt_kernel(const float* __restrict__ src,
                                     __bf16* __restrict__ dst, int R, int C) {
    int t = blockIdx.x * blockDim.x + threadIdx.x;
    if (t >= R * C) return;
    int k = t % R;           // output col  (= src row)
    int n = t / R;           // output row  (= src col)
    dst[(size_t)n * R + k] = (__bf16)src[(size_t)k * C + n];
}

// ---------------------------------------------------------------------------
// 2) GEMM: C[M][N] f32 = A[M][K] bf16 (row major) x Bt[N][K] bf16 (row major)
//    block tile 64x128, K-step 32, 8 waves (2m x 4n), wave tile 32x32
//    (4 WMMAs / 8 fragment loads per wave-step, 4 independent acc chains).
//    Global->LDS staging via GLOBAL_LOAD_ASYNC_TO_LDS_B128, double buffered.
// ---------------------------------------------------------------------------
#define GT_M 64
#define GT_N 128
#define GT_K 32
#define LPAD 40   // LDS row stride in bf16 (32 + 8 pad, keeps 16B alignment)

__global__ __launch_bounds__(256) void gemm_nt_bf16(
    const __bf16* __restrict__ A, const __bf16* __restrict__ Bt,
    float* __restrict__ C, int M, int N, int K)
{
    __shared__ __align__(16) __bf16 As[2][GT_M * LPAD];
    __shared__ __align__(16) __bf16 Bs[2][GT_N * LPAD];

    const int tid  = threadIdx.x;
    const int lane = tid & 31;
    const int w    = tid >> 5;
    const int m0   = blockIdx.y * GT_M;
    const int n0   = blockIdx.x * GT_N;
    const int wm   = (w & 1) * 32;       // wave m-offset within block tile
    const int wn   = (w >> 1) * 32;      // wave n-offset within block tile

    // cooperative-load assignment: 16B per thread per sub-tile
    const int arow   = tid >> 2;         // 0..63
    const int achunk = (tid & 3) * 8;    // bf16 units: 0,8,16,24

    v8f c00 = {0,0,0,0,0,0,0,0}, c01 = {0,0,0,0,0,0,0,0};
    v8f c10 = {0,0,0,0,0,0,0,0}, c11 = {0,0,0,0,0,0,0,0};

    auto issue_tile = [&](int k0, int buf) {
        // A tile: 64 rows x 32 bf16
        async_ld_b128(lds_addr_of(&As[buf][arow * LPAD + achunk]),
                      &A[(size_t)(m0 + arow) * K + k0 + achunk]);
        // B tile: 128 rows x 32 bf16 (two rounds)
        async_ld_b128(lds_addr_of(&Bs[buf][arow * LPAD + achunk]),
                      &Bt[(size_t)(n0 + arow) * K + k0 + achunk]);
        async_ld_b128(lds_addr_of(&Bs[buf][(arow + 64) * LPAD + achunk]),
                      &Bt[(size_t)(n0 + arow + 64) * K + k0 + achunk]);
    };

    const int niter = K / GT_K;
    issue_tile(0, 0);

    const int am = wm + (lane & 15);
    const int bn = wn + (lane & 15);
    const int kg = (lane >> 4) * 8;      // A-fragment K half offset
    const int kb = (lane >> 4) * 8;      // B-fragment K half offset

    for (int it = 0; it < niter; ++it) {
        const int cur = it & 1;
        if (it + 1 < niter) {
            issue_tile((it + 1) * GT_K, cur ^ 1);
            asm volatile("s_wait_asynccnt 0x3" ::: "memory");  // tile `it` done
        } else {
            asm volatile("s_wait_asynccnt 0x0" ::: "memory");
        }
        __syncthreads();   // tile `it` visible to all waves

        const __bf16* Ab = As[cur];
        const __bf16* Bb = Bs[cur];
        bfrag a0 = load_frag2(&Ab[am * LPAD + kg],        &Ab[am * LPAD + 16 + kg]);
        bfrag a1 = load_frag2(&Ab[(am + 16) * LPAD + kg], &Ab[(am + 16) * LPAD + 16 + kg]);
        bfrag b0 = load_frag2(&Bb[bn * LPAD + kb],        &Bb[bn * LPAD + kb + 16]);
        bfrag b1 = load_frag2(&Bb[(bn + 16) * LPAD + kb], &Bb[(bn + 16) * LPAD + kb + 16]);

        c00 = __builtin_amdgcn_wmma_f32_16x16x32_bf16(false, a0, false, b0, (short)0, c00, false, false);
        c01 = __builtin_amdgcn_wmma_f32_16x16x32_bf16(false, a0, false, b1, (short)0, c01, false, false);
        c10 = __builtin_amdgcn_wmma_f32_16x16x32_bf16(false, a1, false, b0, (short)0, c10, false, false);
        c11 = __builtin_amdgcn_wmma_f32_16x16x32_bf16(false, a1, false, b1, (short)0, c11, false, false);

        __syncthreads();   // done reading buf `cur` before tile it+2 overwrites it
    }

    // C layout: lane = N, VGPR r = M row (+8 for upper half-wave)
    const int cn = n0 + wn + (lane & 15);
    const int r0 = m0 + wm + ((lane >> 4) << 3);
    #pragma unroll
    for (int r = 0; r < 8; ++r) {
        C[(size_t)(r0 + r) * N + cn]           = c00[r];
        C[(size_t)(r0 + r) * N + cn + 16]      = c01[r];
        C[(size_t)(r0 + 16 + r) * N + cn]      = c10[r];
        C[(size_t)(r0 + 16 + r) * N + cn + 16] = c11[r];
    }
}

// ---------------------------------------------------------------------------
// 3) RoPE + convert: src f32 [S][nh*D]  ->  dst bf16 [nh][S][D]
// ---------------------------------------------------------------------------
__global__ void rope_cvt_kernel(const float* __restrict__ src,
                                __bf16* __restrict__ dst, int nh) {
    int t = blockIdx.x * blockDim.x + threadIdx.x;
    if (t >= S_LEN * nh * 64) return;
    int i  = t & 63;
    int hh = (t >> 6) % nh;
    int s  = t / (64 * nh);
    const float* row = src + (size_t)s * nh * HD + hh * HD;
    float x1 = row[i];
    float x2 = row[i + 64];
    float invf = __expf(-(float)i * 0.14391157f);   // 10000^(-i/64)
    float fr = (float)s * invf;
    float c = __cosf(fr), sn = __sinf(fr);
    __bf16* drow = dst + ((size_t)hh * S_LEN + s) * HD;
    drow[i]      = (__bf16)(x1 * c - x2 * sn);
    drow[i + 64] = (__bf16)(x2 * c + x1 * sn);
}

// V f32 [S][KVH*D]  ->  Vt bf16 [KVH][D][S]
__global__ void v_transpose_kernel(const float* __restrict__ v,
                                   __bf16* __restrict__ vt) {
    int t = blockIdx.x * blockDim.x + threadIdx.x;
    if (t >= KVH * HD * S_LEN) return;
    int s  = t & (S_LEN - 1);
    int d  = (t >> 11) & (HD - 1);
    int hh = t >> 18;
    vt[t] = (__bf16)(v[(size_t)s * KVHD + hh * HD + d]);
}

// ---------------------------------------------------------------------------
// 4) Sparse mask -> bitmask [H*S][64] u32.  One wave per (h,q) row.
//    selected = cumsum(attendable in importance order) <= K_adjusted
//    HBM-bound: streams 256MB of sorted_indices; early-exit on monotone cumsum.
// ---------------------------------------------------------------------------
__global__ __launch_bounds__(256) void mask_kernel(
    const int* __restrict__ sorted, unsigned int* __restrict__ bitmask)
{
    __shared__ unsigned int sm[8][64];
    const int lane = threadIdx.x & 31;
    const int w    = threadIdx.x >> 5;
    const int row  = blockIdx.x * 8 + w;       // h*S + q
    const int q    = row & (S_LEN - 1);

    sm[w][lane]      = 0u;
    sm[w][lane + 32] = 0u;

    int Kadj = ((q + 4) >> 2) - (MINIDX + WIN);   // ceil((q+1)/4) - 132
    if (Kadj < 0) Kadj = 0;

    if (Kadj > 0) {
        const int* srow = sorted + (size_t)row * S_LEN;
        int running = 0;
        for (int t0 = 0; t0 < S_LEN; t0 += 32) {
            if (t0 + 512 < S_LEN)
                __builtin_prefetch(&srow[t0 + 512 + lane], 0, 1);
            int idx = srow[t0 + lane];
            bool att = (idx <= q);                       // attendable (causal)
            unsigned int bal = __builtin_amdgcn_ballot_w32(att);
            int cum = running + __builtin_popcount(bal & ((1u << lane) - 1u)) + (att ? 1 : 0);
            if (cum <= Kadj)
                atomicOr(&sm[w][idx >> 5], 1u << (idx & 31));
            running += __builtin_popcount(bal);
            if (running >= Kadj) break;                  // wave-uniform early exit
        }
    }
    asm volatile("s_wait_dscnt 0" ::: "memory");
    unsigned int* dst = bitmask + (size_t)row * 64;
    dst[lane]      = sm[w][lane];
    dst[lane + 32] = sm[w][lane + 32];
}

// ---------------------------------------------------------------------------
// 5) Flash attention, one wave per 16-query tile, 32 keys per step.
// ---------------------------------------------------------------------------
__device__ __forceinline__ float attn_bias(const unsigned int* __restrict__ bm_row,
                                           int q, int p) {
    if (p > q) return -INFF;                       // causal
    if (p < MINIDX || p + WIN > q) return 0.0f;    // guaranteed keys + sliding window
    return ((bm_row[p >> 5] >> (p & 31)) & 1u) ? 0.0f : -INFF;
}

__global__ __launch_bounds__(128) void attn_kernel(
    const __bf16* __restrict__ Qb,   // [H][S][D]
    const __bf16* __restrict__ Kb,   // [KVH][S][D]
    const __bf16* __restrict__ Vt,   // [KVH][D][S]
    const unsigned int* __restrict__ bitmask, // [H*S][64]
    __bf16* __restrict__ Ob)         // [S][H*D]
{
    __shared__ __align__(16) __bf16 Pl[4][16 * 32];

    const int lane = threadIdx.x & 31;
    const int w    = threadIdx.x >> 5;
    const int tile = blockIdx.x * 4 + w;
    const int h    = tile >> 7;            // /(S/16)
    const int q0   = (tile & 127) << 4;
    const int kvh  = h >> 1;               // GQA: n_rep = 2

    const __bf16* Qh = Qb + ((size_t)h * S_LEN + q0) * HD;
    const __bf16* Kh = Kb + (size_t)kvh * S_LEN * HD;
    const __bf16* Vh = Vt + (size_t)kvh * HD * S_LEN;

    const int mrow = lane & 15;
    const int hi   = lane >> 4;
    const int kg   = hi * 16;

    // Q tile as 4 A-fragments (d-slices of 32)
    bfrag qf[4];
    #pragma unroll
    for (int j = 0; j < 4; ++j) {
        const __bf16* base = Qh + (size_t)mrow * HD + 32 * j + kg;
        qf[j] = load_frag2(base, base + 8);
    }

    float mi[8], li[8];
    v8f   Ot[8];
    #pragma unroll
    for (int r = 0; r < 8; ++r) { mi[r] = -INFF; li[r] = 0.0f; }
    #pragma unroll
    for (int jo = 0; jo < 8; ++jo) Ot[jo] = (v8f){0,0,0,0,0,0,0,0};

    const float scale = 0.08838834764831845f;  // 1/sqrt(128)
    const int nstep = ((q0 + 15) >> 5) + 1;

    for (int st = 0; st < nstep; ++st) {
        const int kb = st << 5;

        // scores: two 16x16 tiles (keys kb..+15 and kb+16..+31), contract d=128
        v8f s0 = {0,0,0,0,0,0,0,0};
        v8f s1 = {0,0,0,0,0,0,0,0};
        #pragma unroll
        for (int j = 0; j < 4; ++j) {
            const int dbase = 32 * j + hi * 16;
            const __bf16* b0p = Kh + (size_t)(kb      + (lane & 15)) * HD + dbase;
            const __bf16* b1p = Kh + (size_t)(kb + 16 + (lane & 15)) * HD + dbase;
            bfrag b0 = load_frag2(b0p, b0p + 8);
            bfrag b1 = load_frag2(b1p, b1p + 8);
            s0 = __builtin_amdgcn_wmma_f32_16x16x32_bf16(false, qf[j], false, b0, (short)0, s0, false, false);
            s1 = __builtin_amdgcn_wmma_f32_16x16x32_bf16(false, qf[j], false, b1, (short)0, s1, false, false);
        }

        // mask + online softmax (row r lives in lanes of one 16-lane half)
        const int colp = kb + (lane & 15);
        #pragma unroll
        for (int r = 0; r < 8; ++r) {
            const int q = q0 + r + hi * 8;
            const unsigned int* bmr = bitmask + (size_t)(h * S_LEN + q) * 64;
            float v0 = s0[r] * scale + attn_bias(bmr, q, colp);
            float v1 = s1[r] * scale + attn_bias(bmr, q, colp + 16);

            float mx = fmaxf(v0, v1);
            mx = fmaxf(mx, __shfl_xor(mx, 1));
            mx = fmaxf(mx, __shfl_xor(mx, 2));
            mx = fmaxf(mx, __shfl_xor(mx, 4));
            mx = fmaxf(mx, __shfl_xor(mx, 8));
            float mnew = fmaxf(mi[r], mx);
            float mu   = (mnew == -INFF) ? 0.0f : mnew;   // avoid inf-inf NaN
            float alpha = __expf(mi[r] - mu);

            float p0 = __expf(v0 - mu);
            float p1 = __expf(v1 - mu);
            float rs = p0 + p1;
            rs += __shfl_xor(rs, 1);
            rs += __shfl_xor(rs, 2);
            rs += __shfl_xor(rs, 4);
            rs += __shfl_xor(rs, 8);

            li[r] = li[r] * alpha + rs;
            mi[r] = mnew;
            #pragma unroll
            for (int jo = 0; jo < 8; ++jo) Ot[jo][r] *= alpha;

            // P (C-layout) -> LDS 16x32 bf16 for transposition into A-layout
            __bf16* pl = &Pl[w][(r + hi * 8) * 32 + (lane & 15)];
            pl[0]  = (__bf16)p0;
            pl[16] = (__bf16)p1;
        }
        asm volatile("s_wait_dscnt 0" ::: "memory");

        // P as A-fragment (16 queries x 32 keys)
        const __bf16* pbase = &Pl[w][mrow * 32 + kg];
        bfrag pf = load_frag2(pbase, pbase + 8);

        // O += P x V : 8 output d-slices; V rows contiguous in S thanks to Vt
        #pragma unroll
        for (int jo = 0; jo < 8; ++jo) {
            const __bf16* vp = Vh + (size_t)(jo * 16 + (lane & 15)) * S_LEN + kb + hi * 16;
            bfrag vf = load_frag2(vp, vp + 8);
            Ot[jo] = __builtin_amdgcn_wmma_f32_16x16x32_bf16(false, pf, false, vf, (short)0, Ot[jo], false, false);
        }
    }

    // epilogue: O / l, store bf16 to [S][H*D] for the Wo GEMM
    #pragma unroll
    for (int r = 0; r < 8; ++r) {
        const int q = q0 + r + hi * 8;
        const float inv = 1.0f / li[r];   // window guarantees li > 0
        #pragma unroll
        for (int jo = 0; jo < 8; ++jo) {
            const int d = jo * 16 + (lane & 15);
            Ob[(size_t)q * HHD + h * HD + d] = (__bf16)(Ot[jo][r] * inv);
        }
    }
}

// ---------------------------------------------------------------------------
// launcher
// ---------------------------------------------------------------------------
extern "C" void kernel_launch(void* const* d_in, const int* in_sizes, int n_in,
                              void* d_out, int out_size, void* d_ws, size_t ws_size,
                              hipStream_t stream) {
    const float* hidden = (const float*)d_in[0];   // [S][HID]
    const float* Wq     = (const float*)d_in[1];   // [HID][H*D]
    const float* Wk     = (const float*)d_in[2];   // [HID][KVH*D]
    const float* Wv     = (const float*)d_in[3];   // [HID][KVH*D]
    const float* Wo     = (const float*)d_in[4];   // [H*D][HID]
    const int*   sorted = (const int*)d_in[5];     // [1][H][S][S]
    float* out = (float*)d_out;                    // [S][HID]

    char* ws = (char*)d_ws;
    auto take = [&](size_t bytes) { char* p = ws; ws += bytes; return p; };

    __bf16* hid_bf = (__bf16*)take((size_t)S_LEN * HID * 2);
    __bf16* wqT    = (__bf16*)take((size_t)HHD  * HID * 2);
    __bf16* wkT    = (__bf16*)take((size_t)KVHD * HID * 2);
    __bf16* wvT    = (__bf16*)take((size_t)KVHD * HID * 2);
    __bf16* woT    = (__bf16*)take((size_t)HID  * HHD * 2);
    float*  q32    = (float*)take((size_t)S_LEN * HHD  * 4);
    float*  k32    = (float*)take((size_t)S_LEN * KVHD * 4);
    float*  v32    = (float*)take((size_t)S_LEN * KVHD * 4);
    __bf16* qbf    = (__bf16*)take((size_t)NH  * S_LEN * HD * 2);
    __bf16* kbf    = (__bf16*)take((size_t)KVH * S_LEN * HD * 2);
    __bf16* vtbf   = (__bf16*)take((size_t)KVH * HD * S_LEN * 2);
    __bf16* attnbf = (__bf16*)take((size_t)S_LEN * HHD * 2);
    unsigned int* bm = (unsigned int*)take((size_t)NH * S_LEN * 64 * 4);

    const int T = 256;

    // 1) precision conversion / weight transposes
    cvt_bf16_kernel<<<(S_LEN * HID + T - 1) / T, T, 0, stream>>>(hidden, hid_bf, S_LEN * HID);
    transpose_cvt_kernel<<<(HID * HHD  + T - 1) / T, T, 0, stream>>>(Wq, wqT, HID, HHD);
    transpose_cvt_kernel<<<(HID * KVHD + T - 1) / T, T, 0, stream>>>(Wk, wkT, HID, KVHD);
    transpose_cvt_kernel<<<(HID * KVHD + T - 1) / T, T, 0, stream>>>(Wv, wvT, HID, KVHD);
    transpose_cvt_kernel<<<(HHD * HID  + T - 1) / T, T, 0, stream>>>(Wo, woT, HHD, HID);

    // 2) QKV projections (WMMA bf16, async-to-LDS double buffered)
    gemm_nt_bf16<<<dim3(HHD / GT_N,  S_LEN / GT_M), T, 0, stream>>>(hid_bf, wqT, q32, S_LEN, HHD,  HID);
    gemm_nt_bf16<<<dim3(KVHD / GT_N, S_LEN / GT_M), T, 0, stream>>>(hid_bf, wkT, k32, S_LEN, KVHD, HID);
    gemm_nt_bf16<<<dim3(KVHD / GT_N, S_LEN / GT_M), T, 0, stream>>>(hid_bf, wvT, v32, S_LEN, KVHD, HID);

    // 3) RoPE + layout repack
    rope_cvt_kernel<<<(S_LEN * NH  * 64 + T - 1) / T, T, 0, stream>>>(q32, qbf, NH);
    rope_cvt_kernel<<<(S_LEN * KVH * 64 + T - 1) / T, T, 0, stream>>>(k32, kbf, KVH);
    v_transpose_kernel<<<(KVH * HD * S_LEN + T - 1) / T, T, 0, stream>>>(v32, vtbf);

    // 4) sparse mask scan (HBM-bound: 256MB of sorted_indices)
    mask_kernel<<<(NH * S_LEN) / 8, T, 0, stream>>>(sorted, bm);

    // 5) flash attention (WMMA bf16, one wave per 16-query tile)
    attn_kernel<<<(NH * (S_LEN / 16)) / 4, 128, 0, stream>>>(qbf, kbf, vtbf, bm, attnbf);

    // 6) output projection
    gemm_nt_bf16<<<dim3(HID / GT_N, S_LEN / GT_M), T, 0, stream>>>(attnbf, woT, out, S_LEN, HID, HHD);
}
